// BiFormerSTL_25855703122694
// MI455X (gfx1250) — compile-verified, matrix operand order checked by reading, not run on hardware
//
#include <hip/hip_runtime.h>
#include <math.h>

typedef _Float16 f16;
typedef _Float16 v8h  __attribute__((ext_vector_type(8)));
typedef _Float16 v16h __attribute__((ext_vector_type(16)));
typedef float    v8f  __attribute__((ext_vector_type(8)));
typedef float    v4ff __attribute__((ext_vector_type(4)));
typedef float    v8ff __attribute__((ext_vector_type(8)));

#if defined(__has_builtin)
#if __has_builtin(__builtin_amdgcn_global_load_async_to_lds_b128) && \
    __has_builtin(__builtin_amdgcn_s_wait_asynccnt)
#define USE_ASYNC_LDS 1
#endif
#endif
#ifndef USE_ASYNC_LDS
#define USE_ASYNC_LDS 0
#endif

#if USE_ASYNC_LDS
typedef int v4i_ __attribute__((vector_size(16)));
#define ASG __attribute__((address_space(1)))
#define ASL __attribute__((address_space(3)))
#define GPTR16(p) ((ASG v4i_*)(unsigned long long)(p))
#define LPTR16(p) ((ASL v4i_*)(unsigned)(unsigned long long)(p))
#endif

// ---------------------------------------------------------------------------
// Stage one 64x32 A tile (f32 -> f16) and one 64x32 B tile (pre-converted f16,
// N-major rows of length K) into LDS. 128 threads. K % 32 == 0 guaranteed.
// B rows are padded to a multiple of 64 in N, so no N bounds checks anywhere.
// ---------------------------------------------------------------------------
__device__ __forceinline__ void stage_tiles(const float* __restrict__ A,
                                            const f16* __restrict__ Bt,
                                            f16* As, f16* Bs,
                                            int t, int m0, int n0,
                                            int M, int K, int k0)
{
  // ---- A tile: thread t handles row r = t>>1, 16 consecutive k ----
  {
    const int r  = t >> 1;
    const int kh = (t & 1) * 16;
    const int gm = m0 + r;
    f16* dst = As + r * 32 + kh;
    if (gm < M) {
      const v4ff* s4 = (const v4ff*)(A + (long long)gm * K + (k0 + kh));
      v4ff a0 = s4[0], a1 = s4[1], a2 = s4[2], a3 = s4[3];
      v8ff f0 = __builtin_shufflevector(a0, a1, 0, 1, 2, 3, 4, 5, 6, 7);
      v8ff f1 = __builtin_shufflevector(a2, a3, 0, 1, 2, 3, 4, 5, 6, 7);
      *(v8h*)dst       = __builtin_convertvector(f0, v8h);
      *(v8h*)(dst + 8) = __builtin_convertvector(f1, v8h);
    } else {
      v8h z = {};
      *(v8h*)dst = z;
      *(v8h*)(dst + 8) = z;
    }
  }
  // ---- B tile: thread t handles col n = t>>1, 16 consecutive k (pure copy) --
  {
    const int n  = t >> 1;
    const int kh = (t & 1) * 16;
    const f16* src = Bt + (long long)(n0 + n) * K + (k0 + kh);
    f16* dst = Bs + n * 32 + kh;
#if USE_ASYNC_LDS
    __builtin_amdgcn_global_load_async_to_lds_b128(GPTR16(src), LPTR16(dst), 0, 0);
    __builtin_amdgcn_global_load_async_to_lds_b128(GPTR16(src + 8), LPTR16(dst + 8), 0, 0);
#else
    *(v8h*)dst       = *(const v8h*)src;
    *(v8h*)(dst + 8) = *(const v8h*)(src + 8);
#endif
  }
}

// ---------------------------------------------------------------------------
// WMMA GEMM: C[M,N] = A[M,K] @ B[K,N] (+bias) (+exact GELU)
// A f32 row-major; B supplied pre-transposed/converted as f16 [Npad][K].
// Block = 128 threads (4 waves), tile 64x64, K-step 32, double-buffered LDS.
// epilogue: 0 = none, 1 = +bias, 2 = +bias then GELU(exact erf)
// ---------------------------------------------------------------------------
__global__ __launch_bounds__(128)
void gemm_wmma_f16(const float* __restrict__ A, const f16* __restrict__ Bt,
                   const float* __restrict__ bias, float* __restrict__ C,
                   int M, int N, int K, int epilogue)
{
  __shared__ __align__(16) f16 As[2][64 * 32];
  __shared__ __align__(16) f16 Bs[2][64 * 32];

  const int m0 = blockIdx.y * 64;
  const int n0 = blockIdx.x * 64;
  const int t    = threadIdx.x;
  const int lane = t & 31;
  const int wid  = t >> 5;
  const int wr   = wid >> 1;
  const int wc   = wid & 1;
  const int lmod  = lane & 15;
  const int lhalf = lane >> 4;      // 0: K 0-7 & 16-23, 1: K 8-15 & 24-31
  const int kbase = lhalf * 8;

  v8f acc[2][2];
  for (int i = 0; i < 2; ++i)
    for (int j = 0; j < 2; ++j)
      acc[i][j] = v8f{};

  stage_tiles(A, Bt, As[0], Bs[0], t, m0, n0, M, K, 0);
#if USE_ASYNC_LDS
  __builtin_amdgcn_s_wait_asynccnt(0);
#endif
  __syncthreads();

  int cur = 0;
  for (int k0 = 0; k0 < K; k0 += 32) {
    if (k0 + 32 < K)
      stage_tiles(A, Bt, As[cur ^ 1], Bs[cur ^ 1], t, m0, n0, M, K, k0 + 32);

    v16h afrag[2], bfrag[2];
    #pragma unroll
    for (int ti = 0; ti < 2; ++ti) {
      const int row = wr * 32 + ti * 16 + lmod;
      v8h lo = *(const v8h*)&As[cur][row * 32 + kbase];
      v8h hi = *(const v8h*)&As[cur][row * 32 + kbase + 16];
      afrag[ti] = __builtin_shufflevector(lo, hi, 0,1,2,3,4,5,6,7,8,9,10,11,12,13,14,15);
    }
    #pragma unroll
    for (int tj = 0; tj < 2; ++tj) {
      const int col = wc * 32 + tj * 16 + lmod;
      v8h lo = *(const v8h*)&Bs[cur][col * 32 + kbase];
      v8h hi = *(const v8h*)&Bs[cur][col * 32 + kbase + 16];
      bfrag[tj] = __builtin_shufflevector(lo, hi, 0,1,2,3,4,5,6,7,8,9,10,11,12,13,14,15);
    }

    #pragma unroll
    for (int ti = 0; ti < 2; ++ti)
      #pragma unroll
      for (int tj = 0; tj < 2; ++tj)
        acc[ti][tj] = __builtin_amdgcn_wmma_f32_16x16x32_f16(
            false, afrag[ti], false, bfrag[tj], (short)0, acc[ti][tj], false, false);

#if USE_ASYNC_LDS
    __builtin_amdgcn_s_wait_asynccnt(0);
#endif
    __syncthreads();
    cur ^= 1;
  }

  // ---- epilogue: C/D layout -> VGPR j is row (j + 8*lhalf), col = lane%16 --
  const bool full = (m0 + 64 <= M) && (n0 + 64 <= N);
  #pragma unroll
  for (int ti = 0; ti < 2; ++ti) {
    #pragma unroll
    for (int tj = 0; tj < 2; ++tj) {
      const int gn = n0 + wc * 32 + tj * 16 + lmod;
      const int mbase = m0 + wr * 32 + ti * 16 + lhalf * 8;
      if (full) {
        const float bv = (epilogue > 0) ? bias[gn] : 0.0f;
        #pragma unroll
        for (int j = 0; j < 8; ++j) {
          float v = acc[ti][tj][j] + bv;
          if (epilogue == 2) v = 0.5f * v * (1.0f + erff(v * 0.70710678118654752f));
          C[(long long)(mbase + j) * N + gn] = v;
        }
      } else {
        if (gn >= N) continue;
        const float bv = (epilogue > 0) ? bias[gn] : 0.0f;
        #pragma unroll
        for (int j = 0; j < 8; ++j) {
          const int gm = mbase + j;
          if (gm < M) {
            float v = acc[ti][tj][j] + bv;
            if (epilogue == 2) v = 0.5f * v * (1.0f + erff(v * 0.70710678118654752f));
            C[(long long)gm * N + gn] = v;
          }
        }
      }
    }
  }
}

// Convert weight W[K][N] (f32) to Wt[Npad][K] (f16), zero-padded in N.
__global__ void k_w_to_f16t(const float* __restrict__ W, f16* __restrict__ Wt,
                            int K, int N, int Npad)
{
  long long total = (long long)Npad * K;
  long long stride = (long long)gridDim.x * blockDim.x;
  for (long long i = blockIdx.x * (long long)blockDim.x + threadIdx.x; i < total; i += stride) {
    int k = (int)(i % K);
    int n = (int)(i / K);
    Wt[i] = (n < N) ? (f16)W[(long long)k * N + n] : (f16)0.0f;
  }
}

// ---------------------------------------------------------------------------
// Elementwise / small kernels (grid-stride)
// ---------------------------------------------------------------------------
__global__ void k_nchw_to_nhwc(const float* __restrict__ in, float* __restrict__ out,
                               int N, int C, int H, int W)
{
  long long total = (long long)N * C * H * W;
  long long stride = (long long)gridDim.x * blockDim.x;
  for (long long i = blockIdx.x * (long long)blockDim.x + threadIdx.x; i < total; i += stride) {
    int c = (int)(i % C); long long t2 = i / C;
    int x = (int)(t2 % W); t2 /= W;
    int y = (int)(t2 % H); int n = (int)(t2 / H);
    out[i] = in[(((long long)n * C + c) * H + y) * W + x];
  }
}

// VALID conv, stride s, weights HWIO (ks,ks,Cin,Cout)
__global__ void k_conv(const float* __restrict__ x, const float* __restrict__ w,
                       const float* __restrict__ b, float* __restrict__ y,
                       int N, int Hi, int Wi, int Cin, int Ho, int Wo, int Cout,
                       int ks, int s)
{
  long long total = (long long)N * Ho * Wo * Cout;
  long long stride = (long long)gridDim.x * blockDim.x;
  for (long long i = blockIdx.x * (long long)blockDim.x + threadIdx.x; i < total; i += stride) {
    int c = (int)(i % Cout); long long t2 = i / Cout;
    int xo = (int)(t2 % Wo); t2 /= Wo;
    int yo = (int)(t2 % Ho); int n = (int)(t2 / Ho);
    float acc = b[c];
    for (int ky = 0; ky < ks; ++ky)
      for (int kx = 0; kx < ks; ++kx) {
        const float* xp = x + (((long long)n * Hi + (yo * s + ky)) * Wi + (xo * s + kx)) * Cin;
        const float* wp = w + ((long long)(ky * ks + kx) * Cin) * Cout + c;
        for (int ci = 0; ci < Cin; ++ci) acc += xp[ci] * wp[(long long)ci * Cout];
      }
    y[i] = acc;
  }
}

__global__ void k_layernorm(const float* __restrict__ x, const float* __restrict__ g,
                            const float* __restrict__ b, float* __restrict__ y,
                            long long T, int C)
{
  long long stride = (long long)gridDim.x * blockDim.x;
  for (long long t = blockIdx.x * (long long)blockDim.x + threadIdx.x; t < T; t += stride) {
    const float* xp = x + t * C;
    float mu = 0.f;
    for (int c = 0; c < C; ++c) mu += xp[c];
    mu /= (float)C;
    float var = 0.f;
    for (int c = 0; c < C; ++c) { float d = xp[c] - mu; var += d * d; }
    var /= (float)C;
    const float inv = rsqrtf(var + 1e-6f);
    float* yp = y + t * C;
    for (int c = 0; c < C; ++c) yp[c] = (xp[c] - mu) * inv * g[c] + b[c];
  }
}

// window means of q and k over each 7x7 grid window of size h x h
__global__ void k_winmean(const float* __restrict__ qkv, float* __restrict__ wq,
                          float* __restrict__ wk, int N, int h, int C)
{
  const int H = 7 * h, C3 = 3 * C;
  long long total = (long long)N * 49 * C;
  long long stride = (long long)gridDim.x * blockDim.x;
  for (long long i = blockIdx.x * (long long)blockDim.x + threadIdx.x; i < total; i += stride) {
    int c = (int)(i % C); long long t2 = i / C;
    int p = (int)(t2 % 49); int n = (int)(t2 / 49);
    int wy = p / 7, wx = p % 7;
    float sq = 0.f, sk = 0.f;
    for (int py = 0; py < h; ++py)
      for (int px = 0; px < h; ++px) {
        const float* base = qkv + (((long long)n * H + (wy * h + py)) * H + (wx * h + px)) * C3;
        sq += base[c];
        sk += base[C + c];
      }
    const float inv = 1.0f / (float)(h * h);
    wq[i] = sq * inv;
    wk[i] = sk * inv;
  }
}

// per-(n,window) routing logits over 49 windows + top-k selection
__global__ void k_topk(const float* __restrict__ wq, const float* __restrict__ wk,
                       int* __restrict__ ridx, int N, int C, int topk)
{
  long long total = (long long)N * 49;
  long long stride = (long long)gridDim.x * blockDim.x;
  const float scale = rsqrtf((float)C);
  for (long long i = blockIdx.x * (long long)blockDim.x + threadIdx.x; i < total; i += stride) {
    int p = (int)(i % 49); int n = (int)(i / 49);
    const float* qp = wq + ((long long)n * 49 + p) * C;
    float logits[49];
    for (int j = 0; j < 49; ++j) {
      const float* kp = wk + ((long long)n * 49 + j) * C;
      float s = 0.f;
      for (int c = 0; c < C; ++c) s += qp[c] * kp[c];
      logits[j] = s * scale;
    }
    bool used[49];
    for (int j = 0; j < 49; ++j) used[j] = false;
    for (int t2 = 0; t2 < topk; ++t2) {
      int best = -1; float bv = -3.4e38f;
      for (int j = 0; j < 49; ++j)
        if (!used[j] && logits[j] > bv) { bv = logits[j]; best = j; }
      used[best] = true;
      ridx[((long long)n * 49 + p) * topk + t2] = best;
    }
  }
}

// BRA attention: thread per (n, window, query-pixel, head); hd = 32
__global__ void k_bra_attn(const float* __restrict__ qkv, const int* __restrict__ ridx,
                           float* __restrict__ out, int N, int h, int C, int nh, int topk)
{
  const int H = 7 * h, w2 = h * h, C3 = 3 * C;
  const float scale = rsqrtf((float)C);
  long long total = (long long)N * 49 * w2 * nh;
  long long stride = (long long)gridDim.x * blockDim.x;
  for (long long i = blockIdx.x * (long long)blockDim.x + threadIdx.x; i < total; i += stride) {
    int m = (int)(i % nh); long long t2 = i / nh;
    int qpix = (int)(t2 % w2); t2 /= w2;
    int p = (int)(t2 % 49); int n = (int)(t2 / 49);
    const int wy = p / 7, wx = p % 7;
    const int qy = wy * h + qpix / h, qx = wx * h + qpix % h;

    const float* qp = qkv + (((long long)n * H + qy) * H + qx) * C3 + m * 32;
    float q[32];
    #pragma unroll
    for (int d = 0; d < 32; ++d) q[d] = qp[d] * scale;

    float mx = -3.4e38f, se = 0.f;
    float acc[32];
    #pragma unroll
    for (int d = 0; d < 32; ++d) acc[d] = 0.f;

    for (int j = 0; j < topk; ++j) {
      const int wsel = ridx[((long long)n * 49 + p) * topk + j];
      const int sy = (wsel / 7) * h, sx = (wsel % 7) * h;
      for (int pix = 0; pix < w2; ++pix) {
        const int ky = sy + pix / h, kx = sx + pix % h;
        const float* kp = qkv + (((long long)n * H + ky) * H + kx) * C3 + C + m * 32;
        const float* vp = kp + C;
        float s = 0.f;
        #pragma unroll
        for (int d = 0; d < 32; ++d) s += q[d] * kp[d];
        const float nm = fmaxf(mx, s);
        const float eo = __expf(mx - nm);
        const float es = __expf(s - nm);
        se = se * eo + es;
        #pragma unroll
        for (int d = 0; d < 32; ++d) acc[d] = acc[d] * eo + es * vp[d];
        mx = nm;
      }
    }
    const float inv = 1.0f / se;
    float* op = out + (((long long)n * H + qy) * H + qx) * C + m * 32;
    #pragma unroll
    for (int d = 0; d < 32; ++d) op[d] = acc[d] * inv;
  }
}

// global attention (stage 3): thread per (n, query, head); hd = 96, T tokens
__global__ void k_attn_global(const float* __restrict__ qkv, float* __restrict__ out,
                              int N, int T, int C, int nh)
{
  const int hd = 96;             // C / nh for stage 3 (768/8)
  const int C3 = 3 * C;
  const float scale = rsqrtf((float)hd);
  long long total = (long long)N * T * nh;
  long long stride = (long long)gridDim.x * blockDim.x;
  for (long long i = blockIdx.x * (long long)blockDim.x + threadIdx.x; i < total; i += stride) {
    int m = (int)(i % nh); long long t2 = i / nh;
    int qt = (int)(t2 % T); int n = (int)(t2 / T);
    const float* qp = qkv + ((long long)n * T + qt) * C3 + m * hd;
    float q[96];
    #pragma unroll
    for (int d = 0; d < hd; ++d) q[d] = qp[d] * scale;
    float mx = -3.4e38f, se = 0.f;
    float acc[96];
    #pragma unroll
    for (int d = 0; d < hd; ++d) acc[d] = 0.f;
    for (int kt = 0; kt < T; ++kt) {
      const float* kp = qkv + ((long long)n * T + kt) * C3 + C + m * hd;
      const float* vp = kp + C;
      float s = 0.f;
      #pragma unroll
      for (int d = 0; d < hd; ++d) s += q[d] * kp[d];
      const float nm = fmaxf(mx, s);
      const float eo = __expf(mx - nm);
      const float es = __expf(s - nm);
      se = se * eo + es;
      #pragma unroll
      for (int d = 0; d < hd; ++d) acc[d] = acc[d] * eo + es * vp[d];
      mx = nm;
    }
    const float inv = 1.0f / se;
    float* op = out + ((long long)n * T + qt) * C + m * hd;
    #pragma unroll
    for (int d = 0; d < hd; ++d) op[d] = acc[d] * inv;
  }
}

// depthwise 5x5 SAME conv on channels [coff, coff+C) of x (row stride cs); out += result
__global__ void k_dwconv_add(const float* __restrict__ x, int cs, int coff,
                             const float* __restrict__ w, const float* __restrict__ b,
                             float* __restrict__ out, int N, int H, int W, int C)
{
  long long total = (long long)N * H * W * C;
  long long stride = (long long)gridDim.x * blockDim.x;
  for (long long i = blockIdx.x * (long long)blockDim.x + threadIdx.x; i < total; i += stride) {
    int c = (int)(i % C); long long t2 = i / C;
    int xx = (int)(t2 % W); t2 /= W;
    int yy = (int)(t2 % H); int n = (int)(t2 / H);
    float acc = b[c];
    for (int ky = -2; ky <= 2; ++ky) {
      const int sy = yy + ky;
      if (sy < 0 || sy >= H) continue;
      for (int kx = -2; kx <= 2; ++kx) {
        const int sx = xx + kx;
        if (sx < 0 || sx >= W) continue;
        acc += x[(((long long)n * H + sy) * W + sx) * cs + coff + c] *
               w[((ky + 2) * 5 + (kx + 2)) * C + c];
      }
    }
    out[i] += acc;
  }
}

__global__ void k_add(float* __restrict__ y, const float* __restrict__ a, long long total)
{
  long long stride = (long long)gridDim.x * blockDim.x;
  for (long long i = blockIdx.x * (long long)blockDim.x + threadIdx.x; i < total; i += stride)
    y[i] += a[i];
}

__global__ void k_avgpool(const float* __restrict__ x, float* __restrict__ y,
                          int N, int HW, int C)
{
  long long total = (long long)N * C;
  long long stride = (long long)gridDim.x * blockDim.x;
  for (long long i = blockIdx.x * (long long)blockDim.x + threadIdx.x; i < total; i += stride) {
    int c = (int)(i % C); int n = (int)(i / C);
    float s = 0.f;
    for (int t2 = 0; t2 < HW; ++t2) s += x[((long long)n * HW + t2) * C + c];
    y[i] = s / (float)HW;
  }
}

// ---------------------------------------------------------------------------
// Host orchestration
// ---------------------------------------------------------------------------
static inline int grid1d(long long total)
{
  long long b = (total + 255) / 256;
  if (b > 1048576) b = 1048576;
  if (b < 1) b = 1;
  return (int)b;
}

extern "C" void kernel_launch(void* const* d_in, const int* in_sizes, int n_in,
                              void* d_out, int out_size, void* d_ws, size_t ws_size,
                              hipStream_t stream)
{
  (void)in_sizes; (void)n_in; (void)out_size; (void)ws_size;
  const int BN = 16;
  const int EMBED[4] = {96, 192, 384, 768};
  const int DEPTH[4] = {2, 2, 6, 2};
  const int TOPK[4]  = {1, 4, 16, -2};
  const int HS[4]    = {56, 28, 14, 7};

  // ---- unpack inputs (JAX pytree order: dict keys sorted alphabetically) ----
  const float* x_nchw = (const float*)d_in[0];
  int pi = 1;
  auto nxt = [&]() { return (const float*)d_in[pi++]; };

  struct Down { const float *conv_b, *conv_w, *ln_b, *ln_g; } down[4];
  for (int i = 0; i < 4; ++i) {
    down[i].conv_b = nxt(); down[i].conv_w = nxt();
    down[i].ln_b   = nxt(); down[i].ln_g   = nxt();
  }
  const float* head_b = nxt();
  const float* head_w = nxt();
  const float* norm_b = nxt();
  const float* norm_g = nxt();

  struct Blk {
    const float *fc1_b, *fc1_w, *fc2_b, *fc2_w, *lepe_b, *lepe_w;
    const float *n1_b, *n1_g, *n2_b, *n2_g;
    const float *qkv_b, *qkv_w, *wo_b, *wo_w, *proj_b, *proj_w;
  } blk[4][6];
  for (int s = 0; s < 4; ++s)
    for (int b = 0; b < DEPTH[s]; ++b) {
      Blk& B = blk[s][b];
      B.fc1_b = nxt(); B.fc1_w = nxt(); B.fc2_b = nxt(); B.fc2_w = nxt();
      B.lepe_b = nxt(); B.lepe_w = nxt();
      B.n1_b = nxt(); B.n1_g = nxt(); B.n2_b = nxt(); B.n2_g = nxt();
      if (s < 3) { B.qkv_b = nxt(); B.qkv_w = nxt(); B.wo_b = nxt(); B.wo_w = nxt();
                   B.proj_b = nullptr; B.proj_w = nullptr; }
      else       { B.proj_b = nxt(); B.proj_w = nxt(); B.qkv_w = nxt();
                   B.qkv_b = nullptr; B.wo_b = nullptr; B.wo_w = nullptr; }
    }

  // ---- workspace bump allocator --------------------------------------------
  char* ws = (char*)d_ws;
  size_t off = 0;
  auto alloc = [&](size_t bytes) -> void* {
    void* p = ws + off;
    off += (bytes + 255) & ~(size_t)255;
    return p;
  };

  // ---- pre-transpose/convert all GEMM weights to f16 [Npad][K] -------------
  auto wtrans = [&](const float* W, int K, int N) -> const f16* {
    const int Npad = (N + 63) & ~63;
    f16* p = (f16*)alloc((size_t)Npad * K * sizeof(f16));
    k_w_to_f16t<<<grid1d((long long)Npad * K), 256, 0, stream>>>(W, p, K, N, Npad);
    return p;
  };
  struct BlkT { const f16 *qkv_t, *wo_t, *proj_t, *fc1_t, *fc2_t; } blkt[4][6];
  for (int s = 0; s < 4; ++s) {
    const int C = EMBED[s];
    for (int b = 0; b < DEPTH[s]; ++b) {
      const Blk& B = blk[s][b];
      BlkT& Bt = blkt[s][b];
      Bt.qkv_t = wtrans(B.qkv_w, C, 3 * C);
      if (s < 3) { Bt.wo_t = wtrans(B.wo_w, C, C); Bt.proj_t = nullptr; }
      else       { Bt.proj_t = wtrans(B.proj_w, C, C); Bt.wo_t = nullptr; }
      Bt.fc1_t = wtrans(B.fc1_w, C, 4 * C);
      Bt.fc2_t = wtrans(B.fc2_w, 4 * C, C);
    }
  }
  const f16* head_t = wtrans(head_w, 768, 1000);

  // ---- activation buffers --------------------------------------------------
  const long long T0 = (long long)BN * 56 * 56;       // 50176 tokens, stage 0
  float* Xh   = (float*)alloc((size_t)BN * 224 * 224 * 3 * 4);
  float* Y    = (float*)alloc((size_t)(T0 * 96) * 4);
  float* Abuf = (float*)alloc((size_t)(T0 * 96) * 4);
  float* QKV  = (float*)alloc((size_t)(T0 * 288) * 4);
  float* T1   = (float*)alloc((size_t)(T0 * 384) * 4);   // max MLP hidden
  float* ATT  = (float*)alloc((size_t)(T0 * 96) * 4);
  float* WQ   = (float*)alloc((size_t)(BN * 49 * 384) * 4);
  float* WK   = (float*)alloc((size_t)(BN * 49 * 384) * 4);
  int*   RIDX = (int*)  alloc((size_t)(BN * 49 * 16) * 4);
  float* POOL = (float*)alloc((size_t)(BN * 768) * 4);

  auto gemm = [&](const float* A, const f16* Bt, const float* bias, float* C,
                  int M, int N, int K, int ep) {
    dim3 g((N + 63) / 64, (M + 63) / 64);
    gemm_wmma_f16<<<g, 128, 0, stream>>>(A, Bt, bias, C, M, N, K, ep);
  };

  // ---- stem: NCHW -> NHWC, patch embed (4x4 s4), LN ------------------------
  {
    long long tot = (long long)BN * 3 * 224 * 224;
    k_nchw_to_nhwc<<<grid1d(tot), 256, 0, stream>>>(x_nchw, Xh, BN, 3, 224, 224);
    long long ot = T0 * 96;
    k_conv<<<grid1d(ot), 256, 0, stream>>>(Xh, down[0].conv_w, down[0].conv_b, Y,
                                           BN, 224, 224, 3, 56, 56, 96, 4, 4);
    k_layernorm<<<grid1d(T0), 256, 0, stream>>>(Y, down[0].ln_g, down[0].ln_b, Y, T0, 96);
  }

  for (int s = 0; s < 4; ++s) {
    if (s > 0) {
      const int Hp = HS[s - 1], Cp = EMBED[s - 1];
      const long long Tp = (long long)BN * Hp * Hp;
      k_layernorm<<<grid1d(Tp), 256, 0, stream>>>(Y, down[s].ln_g, down[s].ln_b, Abuf, Tp, Cp);
      const long long ot = (long long)BN * HS[s] * HS[s] * EMBED[s];
      k_conv<<<grid1d(ot), 256, 0, stream>>>(Abuf, down[s].conv_w, down[s].conv_b, Y,
                                             BN, Hp, Hp, Cp, HS[s], HS[s], EMBED[s], 2, 2);
    }
    const int H = HS[s], C = EMBED[s], nh = C / 32, h = H / 7;
    const long long T = (long long)BN * H * H;
    const long long TC = T * C;

    for (int b = 0; b < DEPTH[s]; ++b) {
      const Blk& B = blk[s][b];
      const BlkT& BT = blkt[s][b];
      // ---- attention branch ----
      k_layernorm<<<grid1d(T), 256, 0, stream>>>(Y, B.n1_g, B.n1_b, Abuf, T, C);
      if (s < 3) {
        gemm(Abuf, BT.qkv_t, B.qkv_b, QKV, (int)T, 3 * C, C, 1);
        long long wm = (long long)BN * 49 * C;
        k_winmean<<<grid1d(wm), 256, 0, stream>>>(QKV, WQ, WK, BN, h, C);
        k_topk<<<grid1d((long long)BN * 49), 64, 0, stream>>>(WQ, WK, RIDX, BN, C, TOPK[s]);
        long long at = (long long)BN * 49 * h * h * nh;
        k_bra_attn<<<grid1d(at), 128, 0, stream>>>(QKV, RIDX, ATT, BN, h, C, nh, TOPK[s]);
        k_dwconv_add<<<grid1d(TC), 256, 0, stream>>>(QKV, 3 * C, 2 * C, B.lepe_w, B.lepe_b,
                                                     ATT, BN, H, H, C);
        gemm(ATT, BT.wo_t, B.wo_b, T1, (int)T, C, C, 1);
      } else {
        gemm(Abuf, BT.qkv_t, nullptr, QKV, (int)T, 3 * C, C, 0);
        long long at = (long long)BN * 49 * 8;
        k_attn_global<<<grid1d(at), 64, 0, stream>>>(QKV, ATT, BN, 49, C, 8);
        k_dwconv_add<<<grid1d(TC), 256, 0, stream>>>(Abuf, C, 0, B.lepe_w, B.lepe_b,
                                                     ATT, BN, H, H, C);
        gemm(ATT, BT.proj_t, B.proj_b, T1, (int)T, C, C, 1);
      }
      k_add<<<grid1d(TC), 256, 0, stream>>>(Y, T1, TC);

      // ---- MLP branch ----
      k_layernorm<<<grid1d(T), 256, 0, stream>>>(Y, B.n2_g, B.n2_b, Abuf, T, C);
      gemm(Abuf, BT.fc1_t, B.fc1_b, T1, (int)T, 4 * C, C, 2);   // bias + exact GELU
      gemm(T1, BT.fc2_t, B.fc2_b, Abuf, (int)T, C, 4 * C, 1);
      k_add<<<grid1d(TC), 256, 0, stream>>>(Y, Abuf, TC);
    }
  }

  // ---- final norm, pool, head ---------------------------------------------
  const long long Tf = (long long)BN * 49;
  k_layernorm<<<grid1d(Tf), 256, 0, stream>>>(Y, norm_g, norm_b, Abuf, Tf, 768);
  k_avgpool<<<grid1d((long long)BN * 768), 256, 0, stream>>>(Abuf, POOL, BN, 49, 768);
  gemm(POOL, head_t, head_b, (float*)d_out, BN, 1000, 768, 1);
}
